// EnhancedMathematicalReasoning_37864431681928
// MI455X (gfx1250) — compile-verified, matrix operand order checked by reading, not run on hardware
//
#include <hip/hip_runtime.h>
#include <hip/hip_bf16.h>
#include <math.h>

// ---------------------------------------------------------------------------
// EnhancedMathematicalReasoning for MI455X (gfx1250, wave32, WMMA)
//   B=4, S=2048 -> 8192 tokens; H=2048; E=8
//   11 GEMMs of 8192x2048x2048 => compute-bound => bf16 WMMA, f32 accumulate
// ---------------------------------------------------------------------------

typedef __attribute__((ext_vector_type(16))) __bf16 v16bf;
typedef __attribute__((ext_vector_type(8)))  __bf16 v8bf;
typedef __attribute__((ext_vector_type(8)))  float  v8f;

#define TILE   128     // output tile (M x N)
#define KSTEP  32      // K per WMMA (bf16 16x16x32)
#define LDS_K  40      // 32 k elems + 8 pad (80B row stride, keeps 16B align)

__device__ __forceinline__ __bf16 cvt_bf16(float x)  { return (__bf16)x; }
__device__ __forceinline__ __bf16 cvt_bf16(__bf16 x) { return x; }

__device__ __forceinline__ v8f zero_v8f() {
  v8f z;
#pragma unroll
  for (int i = 0; i < 8; ++i) z[i] = 0.f;
  return z;
}

// Fragment load matching the ISA 16-bit A/B layout:
//   lanes 0-15  : K = {khalf*8 .. +7} then {16+khalf*8 .. +7} with khalf=0
//   lanes 16-31 : same with khalf=1
// LDS tile is stored [row][k] (A rows = M, B rows = N after transpose),
// so each half-fragment is a contiguous 16B ds_read_b128.
__device__ __forceinline__ v16bf ld_frag(const __bf16* rowp, int khalf) {
  v8bf lo = *(const v8bf*)(rowp + khalf * 8);
  v8bf hi = *(const v8bf*)(rowp + 16 + khalf * 8);
  v16bf f;
#pragma unroll
  for (int i = 0; i < 8; ++i) { f[i] = lo[i]; f[i + 8] = hi[i]; }
  return f;
}

__device__ __forceinline__ float gelu_erf(float x) {
  return 0.5f * x * (1.0f + erff(x * 0.70710678118654752f));
}

// ---------------------------------------------------------------------------
// Generic tiled GEMM:  out = epilogue(A[M,K] @ B[K,N] + bias)
//   TA  : float or __bf16 activations (converted to bf16 during LDS staging)
//   EPI : 0 -> erf-GELU, store bf16 ; 1 -> * attention_mask[row], store f32
// Block = 256 thr (8 waves); tile 128x128; wave = 64x32 slab (4x2 fragments).
// ---------------------------------------------------------------------------
template <typename TA, int EPI>
__global__ __launch_bounds__(256)
void wmma_gemm_kernel(const TA* __restrict__ A, const float* __restrict__ B,
                      const float* __restrict__ bias, const float* __restrict__ mask,
                      void* __restrict__ outv, int K, int N)
{
  __shared__ __bf16 ldsA[TILE * LDS_K];
  __shared__ __bf16 ldsB[TILE * LDS_K];

  const int tid   = threadIdx.x;
  const int lane  = tid & 31;
  const int wave  = tid >> 5;
  const int wm    = wave >> 2;          // 0..1 : 64-row slab
  const int wn    = wave & 3;           // 0..3 : 32-col slab
  const int bm    = blockIdx.y * TILE;
  const int bn    = blockIdx.x * TILE;
  const int khalf = lane >> 4;
  const int l15   = lane & 15;

  // staging coordinates
  const int arow = tid >> 1;            // 0..127
  const int akb  = (tid & 1) * 16;      // 0 or 16
  const int bk   = tid >> 3;            // 0..31
  const int bnb  = (tid & 7) * 16;      // 0..112

  v8f c[4][2];
#pragma unroll
  for (int mi = 0; mi < 4; ++mi)
#pragma unroll
    for (int ni = 0; ni < 2; ++ni) c[mi][ni] = zero_v8f();

  for (int k0 = 0; k0 < K; k0 += KSTEP) {
    // stage A tile [128 x 32] -> bf16 LDS (row-major)
    {
      const TA* src = A + (size_t)(bm + arow) * K + k0 + akb;
      __bf16* dst = ldsA + arow * LDS_K + akb;
#pragma unroll
      for (int i = 0; i < 16; ++i) dst[i] = cvt_bf16(src[i]);
      if (k0 + KSTEP < K) __builtin_prefetch(src + KSTEP, 0, 0);   // global_prefetch_b8
    }
    // stage B tile [32 x 128] -> transposed bf16 LDS [n][k]
    {
      const float* src = B + (size_t)(k0 + bk) * N + bn + bnb;
#pragma unroll
      for (int i = 0; i < 16; ++i) ldsB[(bnb + i) * LDS_K + bk] = cvt_bf16(src[i]);
      if (k0 + KSTEP < K) __builtin_prefetch(src + (size_t)KSTEP * N, 0, 0);
    }
    __syncthreads();

    v16bf afr[4], bfr[2];
#pragma unroll
    for (int mi = 0; mi < 4; ++mi)
      afr[mi] = ld_frag(ldsA + (wm * 64 + mi * 16 + l15) * LDS_K, khalf);
#pragma unroll
    for (int ni = 0; ni < 2; ++ni)
      bfr[ni] = ld_frag(ldsB + (wn * 32 + ni * 16 + l15) * LDS_K, khalf);

#pragma unroll
    for (int mi = 0; mi < 4; ++mi)
#pragma unroll
      for (int ni = 0; ni < 2; ++ni)
        c[mi][ni] = __builtin_amdgcn_wmma_f32_16x16x32_bf16(
            false, afr[mi], false, bfr[ni], (short)0, c[mi][ni], false, false);
    __syncthreads();
  }

  // epilogue (C layout: VGPR r -> M = r + 8*(lane>=16), N = lane&15)
#pragma unroll
  for (int mi = 0; mi < 4; ++mi) {
#pragma unroll
    for (int ni = 0; ni < 2; ++ni) {
      const int gn = bn + wn * 32 + ni * 16 + l15;
      const float bv = bias[gn];
#pragma unroll
      for (int r = 0; r < 8; ++r) {
        const int gm = bm + wm * 64 + mi * 16 + khalf * 8 + r;
        float v = c[mi][ni][r] + bv;
        if (EPI == 0) {
          ((__bf16*)outv)[(size_t)gm * N + gn] = cvt_bf16(gelu_erf(v));
        } else {
          ((float*)outv)[(size_t)gm * N + gn] = v * mask[gm];
        }
      }
    }
  }
}

// ---------------------------------------------------------------------------
// Router: logits = t1[N,H] @ Wd2[H,8] + bd2 ; op_w = softmax(logits)
// One wave32 per token; lane-strided dot over H, shuffle reduction.
// ---------------------------------------------------------------------------
__global__ __launch_bounds__(256)
void router_kernel(const __bf16* __restrict__ t1, const float* __restrict__ Wd2,
                   const float* __restrict__ bd2, float* __restrict__ opw, int H)
{
  const int token = blockIdx.x * 8 + (threadIdx.x >> 5);
  const int lane  = threadIdx.x & 31;
  const __bf16* row = t1 + (size_t)token * H;

  float acc[8];
#pragma unroll
  for (int e = 0; e < 8; ++e) acc[e] = 0.f;

  for (int k = lane; k < H; k += 32) {
    const float a = (float)row[k];
    const float4 w0 = *(const float4*)(Wd2 + (size_t)k * 8);
    const float4 w1 = *(const float4*)(Wd2 + (size_t)k * 8 + 4);
    acc[0] += a * w0.x; acc[1] += a * w0.y; acc[2] += a * w0.z; acc[3] += a * w0.w;
    acc[4] += a * w1.x; acc[5] += a * w1.y; acc[6] += a * w1.z; acc[7] += a * w1.w;
  }
#pragma unroll
  for (int e = 0; e < 8; ++e)
#pragma unroll
    for (int off = 16; off > 0; off >>= 1)
      acc[e] += __shfl_down(acc[e], off, 32);

  if (lane == 0) {
    float l[8], mx = -3.4e38f;
#pragma unroll
    for (int e = 0; e < 8; ++e) { l[e] = acc[e] + bd2[e]; mx = fmaxf(mx, l[e]); }
    float s = 0.f;
#pragma unroll
    for (int e = 0; e < 8; ++e) { l[e] = __expf(l[e] - mx); s += l[e]; }
    const float inv = 1.f / s;
#pragma unroll
    for (int e = 0; e < 8; ++e) opw[(size_t)token * 8 + e] = l[e] * inv;
  }
}

// ---------------------------------------------------------------------------
// Fused MoE: comb[m,n] = sum_e opw[m,e] * (h @ We[e] + be[e])[m,n]   (bf16 out)
// Per output tile: loop experts, WMMA K-loop each, scale-accumulate into a
// persistent f32 register accumulator; router weights staged once in LDS.
// ---------------------------------------------------------------------------
__global__ __launch_bounds__(256)
void expert_moe_kernel(const float* __restrict__ h, const float* __restrict__ We,
                       const float* __restrict__ be, const float* __restrict__ opw,
                       __bf16* __restrict__ comb, int K, int N)
{
  __shared__ __bf16 ldsA[TILE * LDS_K];
  __shared__ __bf16 ldsB[TILE * LDS_K];
  __shared__ float  ldsW[TILE * 8];

  const int tid   = threadIdx.x;
  const int lane  = tid & 31;
  const int wave  = tid >> 5;
  const int wm    = wave >> 2;
  const int wn    = wave & 3;
  const int bm    = blockIdx.y * TILE;
  const int bn    = blockIdx.x * TILE;
  const int khalf = lane >> 4;
  const int l15   = lane & 15;

  const int arow = tid >> 1;
  const int akb  = (tid & 1) * 16;
  const int bk   = tid >> 3;
  const int bnb  = (tid & 7) * 16;

  // router weights for this row tile: 128 tokens x 8 experts
  *(float4*)(ldsW + tid * 4) = *(const float4*)(opw + (size_t)bm * 8 + tid * 4);

  v8f cb[4][2];
#pragma unroll
  for (int mi = 0; mi < 4; ++mi)
#pragma unroll
    for (int ni = 0; ni < 2; ++ni) cb[mi][ni] = zero_v8f();

  for (int e = 0; e < 8; ++e) {
    const float* B = We + (size_t)e * K * N;

    v8f acc[4][2];
#pragma unroll
    for (int mi = 0; mi < 4; ++mi)
#pragma unroll
      for (int ni = 0; ni < 2; ++ni) acc[mi][ni] = zero_v8f();

    for (int k0 = 0; k0 < K; k0 += KSTEP) {
      {
        const float* src = h + (size_t)(bm + arow) * K + k0 + akb;
        __bf16* dst = ldsA + arow * LDS_K + akb;
#pragma unroll
        for (int i = 0; i < 16; ++i) dst[i] = cvt_bf16(src[i]);
        if (k0 + KSTEP < K) __builtin_prefetch(src + KSTEP, 0, 0);
      }
      {
        const float* src = B + (size_t)(k0 + bk) * N + bn + bnb;
#pragma unroll
        for (int i = 0; i < 16; ++i) ldsB[(bnb + i) * LDS_K + bk] = cvt_bf16(src[i]);
        if (k0 + KSTEP < K) __builtin_prefetch(src + (size_t)KSTEP * N, 0, 0);
      }
      __syncthreads();

      v16bf afr[4], bfr[2];
#pragma unroll
      for (int mi = 0; mi < 4; ++mi)
        afr[mi] = ld_frag(ldsA + (wm * 64 + mi * 16 + l15) * LDS_K, khalf);
#pragma unroll
      for (int ni = 0; ni < 2; ++ni)
        bfr[ni] = ld_frag(ldsB + (wn * 32 + ni * 16 + l15) * LDS_K, khalf);

#pragma unroll
      for (int mi = 0; mi < 4; ++mi)
#pragma unroll
        for (int ni = 0; ni < 2; ++ni)
          acc[mi][ni] = __builtin_amdgcn_wmma_f32_16x16x32_bf16(
              false, afr[mi], false, bfr[ni], (short)0, acc[mi][ni], false, false);
      __syncthreads();   // also protects restaging for next expert
    }

    // weighted combine: cb += opw[m,e] * (acc + be[e,n])
#pragma unroll
    for (int mi = 0; mi < 4; ++mi) {
#pragma unroll
      for (int ni = 0; ni < 2; ++ni) {
        const int gn = bn + wn * 32 + ni * 16 + l15;
        const float bv = be[(size_t)e * N + gn];
#pragma unroll
        for (int r = 0; r < 8; ++r) {
          const int ml = wm * 64 + mi * 16 + khalf * 8 + r;   // local row
          cb[mi][ni][r] += ldsW[ml * 8 + e] * (acc[mi][ni][r] + bv);
        }
      }
    }
  }

  // store combined (bf16) for next GEMM stage
#pragma unroll
  for (int mi = 0; mi < 4; ++mi)
#pragma unroll
    for (int ni = 0; ni < 2; ++ni) {
      const int gn = bn + wn * 32 + ni * 16 + l15;
#pragma unroll
      for (int r = 0; r < 8; ++r) {
        const int gm = bm + wm * 64 + mi * 16 + khalf * 8 + r;
        comb[(size_t)gm * N + gn] = cvt_bf16(cb[mi][ni][r]);
      }
    }
}

// ---------------------------------------------------------------------------
extern "C" void kernel_launch(void* const* d_in, const int* in_sizes, int n_in,
                              void* d_out, int out_size, void* d_ws, size_t ws_size,
                              hipStream_t stream)
{
  const int NT = 4 * 2048;   // B*S tokens
  const int H  = 2048;

  const float* hidden = (const float*)d_in[0];
  const float* mask   = (const float*)d_in[1];
  const float* Wd1    = (const float*)d_in[2];
  const float* bd1    = (const float*)d_in[3];
  const float* Wd2    = (const float*)d_in[4];
  const float* bd2    = (const float*)d_in[5];
  const float* We     = (const float*)d_in[6];
  const float* be     = (const float*)d_in[7];
  const float* Wi1    = (const float*)d_in[8];
  const float* bi1    = (const float*)d_in[9];
  const float* Wi2    = (const float*)d_in[10];
  const float* bi2    = (const float*)d_in[11];

  char* ws = (char*)d_ws;
  const size_t actBytes = (size_t)NT * H * sizeof(__bf16);   // 33.5 MB
  __bf16* t1   = (__bf16*)(ws);                 // also reused as t2 (t1 dead after router)
  __bf16* comb = (__bf16*)(ws + actBytes);
  float*  opw  = (float*)(ws + 2 * actBytes);   // NT*8 f32

  dim3 grid(H / TILE, NT / TILE);   // 16 x 64 tiles
  dim3 blk(256);

  // 1) t1 = gelu(h @ Wd1 + bd1)
  wmma_gemm_kernel<float, 0><<<grid, blk, 0, stream>>>(hidden, Wd1, bd1, nullptr, t1, H, H);
  // 2) op_w = softmax(t1 @ Wd2 + bd2)
  router_kernel<<<NT / 8, blk, 0, stream>>>(t1, Wd2, bd2, opw, H);
  // 3) comb = sum_e op_w[:,e] * (h @ We[e] + be[e])
  expert_moe_kernel<<<grid, blk, 0, stream>>>(hidden, We, be, opw, comb, H, H);
  // 4) t2 = gelu(comb @ Wi1 + bi1)          (t2 aliases t1 storage)
  wmma_gemm_kernel<__bf16, 0><<<grid, blk, 0, stream>>>(comb, Wi1, bi1, nullptr, t1, H, H);
  // 5) out = (t2 @ Wi2 + bi2) * mask        (f32 output)
  wmma_gemm_kernel<__bf16, 1><<<grid, blk, 0, stream>>>(t1, Wi2, bi2, mask, d_out, H, H);
}
